// GuidedUpsample_40724879901084
// MI455X (gfx1250) — matrix operand order repaired and checked by reading, not compiled, not run on hardware
//
#include <hip/hip_runtime.h>
#include <hip/hip_bf16.h>

// gfx1250: wave32, WMMA bf16 16x16x32 with f32 accumulate.
typedef __bf16 bf16;
typedef __attribute__((ext_vector_type(16))) __bf16 v16bf;
typedef __attribute__((ext_vector_type(8)))  __bf16 v8bf;
typedef __attribute__((ext_vector_type(8)))  float  v8f;

#define NDIM 48
#define DHW  (48*48*48)       // 110592 voxels
#define CIN  32
#define HIDC 64
#define ODHW (96*96*96)

__device__ __forceinline__ int clampi(int v, int hi) { return v < 0 ? 0 : (v > hi ? hi : v); }

__device__ __forceinline__ bf16 f2bf(float f) {
  unsigned u = __builtin_bit_cast(unsigned, f);
  unsigned r = (u + 0x7FFFu + ((u >> 16) & 1u)) >> 16;   // round-to-nearest-even
  unsigned short us = (unsigned short)r;
  return __builtin_bit_cast(bf16, us);
}

// 16-bit A-matrix 16x32 per-lane K mapping (contiguous, matches B):
// lanes 0-15:  K=e       (K 0..15)
// lanes 16-31: K=e+16    (K 16..31)
__device__ __forceinline__ int kmapA(int l, int e) {
  return (l < 16) ? e : e + 16;
}

// ---------- prep kernels (run every call; deterministic) ----------

// features [32][48^3] f32 -> channel-last bf16 [48^3][32]
__global__ void prep_feat(const float* __restrict__ feat, bf16* __restrict__ fcl) {
  int idx = blockIdx.x * 256 + threadIdx.x;      // DHW*CIN threads
  int c = idx & 31, v = idx >> 5;
  fcl[idx] = f2bf(feat[c * DHW + v]);
}

// w1 [64][32][27] f32 -> A-fragments: frag id (t*4+mt), 512 bf16 each (32 lanes x 16 elems)
__global__ void prep_w1(const float* __restrict__ w1, bf16* __restrict__ aw1) {
  int f = blockIdx.x * 256 + threadIdx.x;        // 27*4*32*16 = 55296
  int e = f & 15, l = (f >> 4) & 31, mt = (f >> 9) & 3, t = f >> 11;
  int M = mt * 16 + (l & 15);
  int K = kmapA(l, e);
  aw1[f] = f2bf(w1[(M * 32 + K) * 27 + t]);
}

// w2 [216][64] f32 -> A-fragments: frag id ((s*2+mt2)*2+kt), taps padded 27->32 with zeros
__global__ void prep_w2(const float* __restrict__ w2, bf16* __restrict__ aw2) {
  int f = blockIdx.x * 256 + threadIdx.x;        // 8*2*2*32*16 = 32768
  int e = f & 15, l = (f >> 4) & 31, kt = (f >> 9) & 1, mt2 = (f >> 10) & 1, s = f >> 11;
  int tap = mt2 * 16 + (l & 15);
  int K = kt * 32 + kmapA(l, e);
  float v = (tap < 27) ? w2[(s * 27 + tap) * 64 + K] : 0.f;
  aw2[f] = f2bf(v);
}

// ---------- fused main kernel ----------
// 8 waves / block, each wave owns a strip of 16 voxels along W at fixed (z,y).
__global__ __launch_bounds__(256)
void guided_upsample_main(const float* __restrict__ xg,
                          const float* __restrict__ b1,
                          const float* __restrict__ b2,
                          const bf16*  __restrict__ fcl,
                          const bf16*  __restrict__ aw1,
                          const bf16*  __restrict__ aw2,
                          float* __restrict__ out) {
  // per-wave LDS: h[16 vox][64 ch] bf16 (2KB) | m[32 tap][16 vox] f32 (2KB) | wsoft[16][28][2] f32 (3.5KB)
  __shared__ __align__(16) unsigned char smem[8 * 7680];
  const int lane = threadIdx.x & 31;
  const int wave = threadIdx.x >> 5;
  unsigned char* wb = smem + wave * 7680;
  bf16*  hlds = (bf16*)wb;
  float* mlds = (float*)(wb + 2048);
  float* wsft = (float*)(wb + 4096);

  const int sid = blockIdx.x * 8 + wave;         // 6912 strips total, exact
  const int xs = sid % 3;
  const int y  = (sid / 3) % 48;
  const int z  = sid / 144;
  const int x0 = xs * 16;
  const int vox = lane & 15;
  const int hi  = lane >> 4;                     // lane half

  // ---- conv1 (replicate pad): D[64][16] += sum over 27 taps of W1_tap[64][32] x F_tap[32][16]
  // Fully unrolled: A-fragment loads get static immediate offsets from one base,
  // accumulators stay pinned (no rotation movs), scheduler can prefetch across taps.
  v8f c0 = {}, c1 = {}, c2 = {}, c3 = {};
  const bf16* abase1 = aw1 + lane * 16;
  const bf16* fbase  = fcl + hi * 16;
#pragma unroll
  for (int t = 0; t < 27; ++t) {
    const int dz = t / 9 - 1, dy = (t / 3) % 3 - 1, dx = t % 3 - 1;
    const int zc = clampi(z + dz, 47), yc = clampi(y + dy, 47);
    const int xc = clampi(x0 + vox + dx, 47);
    const v16bf bfrag = *(const v16bf*)(fbase + ((zc * 48 + yc) * 48 + xc) * 32);
    const v16bf a0 = *(const v16bf*)(abase1 + (t * 4 + 0) * 512);
    const v16bf a1 = *(const v16bf*)(abase1 + (t * 4 + 1) * 512);
    const v16bf a2 = *(const v16bf*)(abase1 + (t * 4 + 2) * 512);
    const v16bf a3 = *(const v16bf*)(abase1 + (t * 4 + 3) * 512);
    c0 = __builtin_amdgcn_wmma_f32_16x16x32_bf16(false, a0, false, bfrag, (short)0, c0, false, false);
    c1 = __builtin_amdgcn_wmma_f32_16x16x32_bf16(false, a1, false, bfrag, (short)0, c1, false, false);
    c2 = __builtin_amdgcn_wmma_f32_16x16x32_bf16(false, a2, false, bfrag, (short)0, c2, false, false);
    c3 = __builtin_amdgcn_wmma_f32_16x16x32_bf16(false, a3, false, bfrag, (short)0, c3, false, false);
  }

  // ---- bias + SiLU (f32), store h to LDS as bf16 in conv2-B layout [vox][ch]
  v8f accv[4] = {c0, c1, c2, c3};
#pragma unroll
  for (int mt = 0; mt < 4; ++mt) {
    const int ch0 = mt * 16 + hi * 8;            // C-layout: vgpr r -> M = mt*16 + r + hi*8
    const float4 bv0 = *(const float4*)(b1 + ch0);
    const float4 bv1 = *(const float4*)(b1 + ch0 + 4);
    float bb[8] = {bv0.x, bv0.y, bv0.z, bv0.w, bv1.x, bv1.y, bv1.z, bv1.w};
    v8bf hb;
#pragma unroll
    for (int r = 0; r < 8; ++r) {
      float v = accv[mt][r] + bb[r];
      v = v / (1.f + __expf(-v));                // SiLU
      hb[r] = f2bf(v);
    }
    *(v8bf*)(hlds + vox * 64 + ch0) = hb;        // ds_store_b128
  }
  asm volatile("s_wait_dscnt 0" ::: "memory");

  // h B-fragments (K = kt*32 + hi*16 + e, contiguous)
  const v16bf hb0 = *(const v16bf*)(hlds + vox * 64 + 0  + hi * 16);
  const v16bf hb1 = *(const v16bf*)(hlds + vox * 64 + 32 + hi * 16);

  // x neighborhood offsets for this strip (replicate clamp), reused across all subs/channels
  int off27[27];
#pragma unroll
  for (int n = 0; n < 27; ++n) {
    const int dz = n / 9 - 1, dy = (n / 3) % 3 - 1, dx = n % 3 - 1;
    const int zc = clampi(z + dz, 47), yc = clampi(y + dy, 47);
    const int xc = clampi(x0 + vox + dx, 47);
    off27[n] = (zc * 48 + yc) * 48 + xc;
  }
  // warm L2/L0 for the x rows this strip will combine (global_prefetch_b8)
  __builtin_prefetch(xg + (hi * 16) * DHW + off27[4],  0, 1);
  __builtin_prefetch(xg + (hi * 16) * DHW + off27[13], 0, 1);
  __builtin_prefetch(xg + (hi * 16) * DHW + off27[22], 0, 1);

  // ---- per sub-pair: conv2 (WMMA) -> softmax -> 27-tap combine of x
  for (int sp = 0; sp < 4; ++sp) {
    const int si = sp >> 1, sj = sp & 1;         // sub = ((si*2+sj)*2+k), k = s2
#pragma unroll
    for (int s2 = 0; s2 < 2; ++s2) {
      const int s = sp * 2 + s2;
      const bf16* a2p = aw2 + s * 2048 + lane * 16;
      const v16bf af00 = *(const v16bf*)(a2p + 0 * 512);
      const v16bf af01 = *(const v16bf*)(a2p + 1 * 512);
      const v16bf af10 = *(const v16bf*)(a2p + 2 * 512);
      const v16bf af11 = *(const v16bf*)(a2p + 3 * 512);
      v8f m0 = {}, m1 = {};
      m0 = __builtin_amdgcn_wmma_f32_16x16x32_bf16(false, af00, false, hb0, (short)0, m0, false, false);
      m0 = __builtin_amdgcn_wmma_f32_16x16x32_bf16(false, af01, false, hb1, (short)0, m0, false, false);
      m1 = __builtin_amdgcn_wmma_f32_16x16x32_bf16(false, af10, false, hb0, (short)0, m1, false, false);
      m1 = __builtin_amdgcn_wmma_f32_16x16x32_bf16(false, af11, false, hb1, (short)0, m1, false, false);
      // scatter logits (+bias) to LDS [tap][vox]
      v8f mm[2] = {m0, m1};
#pragma unroll
      for (int mt2 = 0; mt2 < 2; ++mt2) {
        const int tap0 = mt2 * 16 + hi * 8;
#pragma unroll
        for (int r = 0; r < 8; ++r) {
          const int tap = tap0 + r;
          const float bv = (tap < 27) ? b2[s * 27 + tap] : 0.f;
          mlds[tap * 16 + vox] = mm[mt2][r] + bv;
        }
      }
      asm volatile("s_wait_dscnt 0" ::: "memory");
      // softmax over 27 taps (both halves compute; low half publishes)
      float mx = -1e30f;
      for (int n = 0; n < 27; ++n) mx = fmaxf(mx, mlds[n * 16 + vox]);
      float den = 0.f;
      for (int n = 0; n < 27; ++n) den += __expf(mlds[n * 16 + vox] - mx);
      const float inv = 1.f / den;
      if (lane < 16) {
        for (int n = 0; n < 27; ++n)
          wsft[(vox * 28 + n) * 2 + s2] = __expf(mlds[n * 16 + vox] - mx) * inv;
      }
      asm volatile("s_wait_dscnt 0" ::: "memory");
    }

    // load this pair's softmax weights once (k=0,1 packed as float2)
    float w0a[27], w1a[27];
#pragma unroll
    for (int n = 0; n < 27; ++n) {
      float2 w = *(const float2*)(wsft + (vox * 28 + n) * 2);
      w0a[n] = w.x; w1a[n] = w.y;
    }

    // combine: out[c, 2z+si, 2y+sj, 2x+k] = sum_n w[n] * x[c, nbr(n)]  (f32)
    for (int c2 = 0; c2 < 16; ++c2) {
      const int c = c2 + hi * 16;
      const float* xp = xg + c * DHW;
      float a0 = 0.f, a1 = 0.f;
#pragma unroll
      for (int n = 0; n < 27; ++n) {
        const float xv = xp[off27[n]];
        a0 = fmaf(w0a[n], xv, a0);
        a1 = fmaf(w1a[n], xv, a1);
      }
      float* op = out + c * ODHW + ((2 * z + si) * 96 + (2 * y + sj)) * 96 + 2 * (x0 + vox);
      *(float2*)op = make_float2(a0, a1);        // coalesced 128B per half-wave
    }
  }
}

extern "C" void kernel_launch(void* const* d_in, const int* in_sizes, int n_in,
                              void* d_out, int out_size, void* d_ws, size_t ws_size,
                              hipStream_t stream) {
  (void)in_sizes; (void)n_in; (void)out_size; (void)ws_size;
  const float* x    = (const float*)d_in[0];
  const float* feat = (const float*)d_in[1];
  const float* w1   = (const float*)d_in[2];
  const float* b1   = (const float*)d_in[3];
  const float* w2   = (const float*)d_in[4];
  const float* b2   = (const float*)d_in[5];
  float* out = (float*)d_out;

  // workspace layout (bytes): fcl 7,077,888 | aw1 110,592 | aw2 65,536
  bf16* fcl = (bf16*)d_ws;
  bf16* aw1 = (bf16*)((char*)d_ws + 7077888);
  bf16* aw2 = (bf16*)((char*)d_ws + 7077888 + 110592);

  prep_feat<<<13824, 256, 0, stream>>>(feat, fcl);   // 48^3 * 32 elems
  prep_w1  <<<  216, 256, 0, stream>>>(w1, aw1);     // 55296 elems
  prep_w2  <<<  128, 256, 0, stream>>>(w2, aw2);     // 32768 elems
  guided_upsample_main<<<864, 256, 0, stream>>>(x, b1, b2, fcl, aw1, aw2, out);
}
